// MambaForecaster_75866302316798
// MI455X (gfx1250) — matrix operand order repaired and checked by reading, not compiled, not run on hardware
//
#include <hip/hip_runtime.h>
#include <hip/hip_bf16.h>
#include <math.h>
#include <stdint.h>

#define B_SZ      32
#define L_SEQ     512
#define DM        512      // D_MODEL
#define NL        4        // N_LAYERS
#define DSTATE    64
#define DCONV     4
#define HEADD     64
#define PRED      96
#define DINNER    1024
#define NH        16       // NHEADS
#define CONVD     1152     // CONV_DIM
#define DPROJ     2192     // D_IN_PROJ
#define EPSV      1e-5f

typedef __bf16 bf16_t;
typedef __attribute__((ext_vector_type(16))) __bf16 v16bf;
typedef __attribute__((ext_vector_type(8)))  float  v8f;

// ---------------------------------------------------------------------------
// Block reduction helper (wave32-aware: shfl width 32, 8 waves per 256 block)
// ---------------------------------------------------------------------------
__device__ inline float blockReduceSum(float v) {
  for (int off = 16; off > 0; off >>= 1) v += __shfl_down(v, off, 32);
  __shared__ float s[8];
  const int lane = threadIdx.x & 31;
  const int w    = threadIdx.x >> 5;
  if (lane == 0) s[w] = v;
  __syncthreads();
  v = (threadIdx.x < 8) ? s[threadIdx.x] : 0.f;
  if (threadIdx.x < 32)
    for (int off = 4; off > 0; off >>= 1) v += __shfl_down(v, off, 32);
  __shared__ float tot;
  if (threadIdx.x == 0) tot = v;
  __syncthreads();
  return tot;
}

// ---------------------------------------------------------------------------
// C[M,N] = A[M,K] @ W[N,K]^T    f32 in/out, bf16 WMMA math (f32 accumulate)
//
// Block = 512-row x 16-col strip (8 waves). All waves share one N-tile so the
// weight fragment addresses are identical across waves (L1 broadcast, weights
// L2-resident). Each wave computes 64x16: one B fragment reused by 4 WMMAs
// with independent accumulators (4 XDL ops in flight per iteration).
//
// Fragment layouts per CDNA5 ISA 7.12.2:
//   A 16x32 bf16 : m = lane&15 ; lanes<16 hold K{0..7,16..23}, lanes>=16 K{8..15,24..31}
//   B 32x16 bf16 : n = lane&15 ; lanes<16 hold K{0..7,16..23}, lanes>=16 K{8..15,24..31}
//   C/D f32      : VGPR r -> M = r + (lane>=16 ? 8:0), N = lane&15
// ---------------------------------------------------------------------------
__global__ __launch_bounds__(256) void gemm_bf16(const float* __restrict__ A,
                                                 const float* __restrict__ W,
                                                 float* __restrict__ C,
                                                 int M, int N, int K) {
  const int tilesN = N >> 4;
  const int strip  = blockIdx.x / tilesN;
  const int n0     = (blockIdx.x % tilesN) << 4;
  const int m0     = strip * 512 + (threadIdx.x >> 5) * 64;
  if (m0 >= M) return;                      // wave-uniform guard
  const int lane = threadIdx.x & 31;
  const int half = lane >> 4;
  const int l15  = lane & 15;
  const float* wrow  = W + (size_t)(n0 + l15) * K;
  const float* arow0 = A + (size_t)(m0 + l15) * K;
  const size_t subStride = (size_t)16 * K;  // 16 rows per subtile
  const int kA = half << 3;                 // 0 / 8
  const int kB = half << 4;                 // 0 / 16
  v8f acc0 = {}, acc1 = {}, acc2 = {}, acc3 = {};
  for (int k0 = 0; k0 < K; k0 += 32) {
    v16bf b;
#pragma unroll
    for (int i = 0; i < 8; ++i) { b[i] = (bf16_t)wrow[k0 + kA + i]; b[8 + i] = (bf16_t)wrow[k0 + 16 + kA + i]; }
    v16bf a0, a1, a2, a3;
#pragma unroll
    for (int i = 0; i < 8; ++i) {
      a0[i]     = (bf16_t)arow0[k0 + kA + i];
      a0[8 + i] = (bf16_t)arow0[k0 + 16 + kA + i];
      a1[i]     = (bf16_t)arow0[subStride + k0 + kA + i];
      a1[8 + i] = (bf16_t)arow0[subStride + k0 + 16 + kA + i];
      a2[i]     = (bf16_t)arow0[2 * subStride + k0 + kA + i];
      a2[8 + i] = (bf16_t)arow0[2 * subStride + k0 + 16 + kA + i];
      a3[i]     = (bf16_t)arow0[3 * subStride + k0 + kA + i];
      a3[8 + i] = (bf16_t)arow0[3 * subStride + k0 + 16 + kA + i];
    }
    if (k0 + 32 < K) {                      // global_prefetch_b8 next K tile
      __builtin_prefetch(arow0 + k0 + 32, 0, 1);
      __builtin_prefetch(wrow + k0 + 32, 0, 1);
    }
    acc0 = __builtin_amdgcn_wmma_f32_16x16x32_bf16(false, a0, false, b, (short)0, acc0, false, false);
    acc1 = __builtin_amdgcn_wmma_f32_16x16x32_bf16(false, a1, false, b, (short)0, acc1, false, false);
    acc2 = __builtin_amdgcn_wmma_f32_16x16x32_bf16(false, a2, false, b, (short)0, acc2, false, false);
    acc3 = __builtin_amdgcn_wmma_f32_16x16x32_bf16(false, a3, false, b, (short)0, acc3, false, false);
  }
  float* crow = C + (size_t)(m0 + (half << 3)) * N + n0 + l15;
  const size_t cSub = (size_t)16 * N;
#pragma unroll
  for (int r = 0; r < 8; ++r) {
    crow[(size_t)r * N]            = acc0[r];
    crow[cSub + (size_t)r * N]     = acc1[r];
    crow[2 * cSub + (size_t)r * N] = acc2[r];
    crow[3 * cSub + (size_t)r * N] = acc3[r];
  }
}

// ---------------------------------------------------------------------------
// Final head: out[b,l',e] = H[b, L-PRED+l', :] @ final_w^T + final_b
// Per-lane row remap (subtiles may straddle batches; each lane maps its own
// row independently, so this is safe).
// ---------------------------------------------------------------------------
__global__ __launch_bounds__(256) void gemm_tail_bf16(const float* __restrict__ H,
                                                      const float* __restrict__ W,
                                                      const float* __restrict__ bias,
                                                      float* __restrict__ C) {
  const int tilesN = DM >> 4;               // 32
  const int strip  = blockIdx.x / tilesN;
  const int n0     = (blockIdx.x % tilesN) << 4;
  const int m0     = strip * 512 + (threadIdx.x >> 5) * 64;
  const int lane = threadIdx.x & 31;
  const int half = lane >> 4;
  const int l15  = lane & 15;
  const float* wrow = W + (size_t)(n0 + l15) * DM;
  const float* arow[4];
#pragma unroll
  for (int s = 0; s < 4; ++s) {
    const int r    = m0 + s * 16 + l15;     // flat row in (B*PRED)
    const int hrow = (r / PRED) * L_SEQ + (L_SEQ - PRED) + (r % PRED);
    arow[s] = H + (size_t)hrow * DM;
  }
  const int kA = half << 3;
  const int kB = half << 4;
  v8f acc[4];
#pragma unroll
  for (int s = 0; s < 4; ++s) acc[s] = (v8f){};
  for (int k0 = 0; k0 < DM; k0 += 32) {
    v16bf b;
#pragma unroll
    for (int i = 0; i < 8; ++i) { b[i] = (bf16_t)wrow[k0 + kA + i]; b[8 + i] = (bf16_t)wrow[k0 + 16 + kA + i]; }
#pragma unroll
    for (int s = 0; s < 4; ++s) {
      v16bf a;
#pragma unroll
      for (int i = 0; i < 8; ++i) {
        a[i]     = (bf16_t)arow[s][k0 + kA + i];
        a[8 + i] = (bf16_t)arow[s][k0 + 16 + kA + i];
      }
      acc[s] = __builtin_amdgcn_wmma_f32_16x16x32_bf16(false, a, false, b,
                                                       (short)0, acc[s], false, false);
    }
  }
  const float bn = bias[n0 + l15];
#pragma unroll
  for (int s = 0; s < 4; ++s)
#pragma unroll
    for (int rr = 0; rr < 8; ++rr)
      C[(size_t)(m0 + s * 16 + (half << 3) + rr) * DM + n0 + l15] = acc[s][rr] + bn;
}

// ---------------------------------------------------------------------------
// dt = softplus(zxbcdt[..., -NH:] + dt_bias)
// ---------------------------------------------------------------------------
__global__ void dt_softplus_kernel(const float* __restrict__ zxbcdt,
                                   const float* __restrict__ dtb,
                                   float* __restrict__ dt) {
  const int idx = blockIdx.x * blockDim.x + threadIdx.x;   // B*L*NH
  const int h  = idx & (NH - 1);
  const int bl = idx >> 4;
  const float v = zxbcdt[(size_t)bl * DPROJ + (DPROJ - NH) + h] + dtb[h];
  dt[idx] = (v > 20.f) ? v : log1pf(__expf(v));
}

// ---------------------------------------------------------------------------
// Causal depthwise conv (width 4) + bias + SiLU on the xBC slice
// ---------------------------------------------------------------------------
__global__ void conv_silu_kernel(const float* __restrict__ zxbcdt,
                                 const float* __restrict__ cw,
                                 const float* __restrict__ cb,
                                 float* __restrict__ xbc) {
  const int idx = blockIdx.x * blockDim.x + threadIdx.x;   // B*L*CONVD
  const int c  = idx % CONVD;
  const int bl = idx / CONVD;
  const int l  = bl % L_SEQ;
  float acc = cb[c];
  const float* col = zxbcdt + (size_t)bl * DPROJ + DINNER + c;
#pragma unroll
  for (int j = 0; j < DCONV; ++j) {
    const int dl = l - (DCONV - 1) + j;
    if (dl >= 0)
      acc += cw[c * DCONV + j] * col[(ptrdiff_t)(dl - l) * DPROJ];
  }
  const float s = 1.f / (1.f + __expf(-acc));
  xbc[idx] = acc * s;
}

// ---------------------------------------------------------------------------
// SSD recurrence. One block per (b,h); 64x64 f32 state lives in registers
// (16 floats per lane, 256 lanes). Per-step x/B/C rows (192 f32, all 16B
// aligned: row stride 4608B, slice offsets 256/4096B) are staged into LDS
// with CDNA5 async-to-LDS copies issued by threads 0-47, published with
// s_wait_asynccnt + workgroup barrier. y reduced across the 4 lanes sharing
// a p with wave32 shfl_xor.
// ---------------------------------------------------------------------------
__global__ __launch_bounds__(256) void ssd_scan_kernel(const float* __restrict__ xbc,
                                                       const float* __restrict__ dt,
                                                       const float* __restrict__ A_log,
                                                       const float* __restrict__ Dp,
                                                       float* __restrict__ y) {
  const int b = blockIdx.x >> 4;
  const int h = blockIdx.x & 15;
  const float A  = -__expf(A_log[h]);
  const float Dh = Dp[h];
  const int tid = threadIdx.x;
  const int p   = tid >> 2;            // 0..63
  const int nb  = (tid & 3) << 4;      // 0,16,32,48
  float hs[16];
#pragma unroll
  for (int j = 0; j < 16; ++j) hs[j] = 0.f;
  __shared__ __align__(16) float stage[192];   // [0:64)=x, [64:128)=B, [128:192)=C
  const uint32_t ldsBase = (uint32_t)(uintptr_t)stage;
  for (int l = 0; l < L_SEQ; ++l) {
    __syncthreads();                   // prev-step LDS reads complete
    const float* row = xbc + (size_t)(b * L_SEQ + l) * CONVD;
    if (tid < 48) {                    // 48 lanes x 16B = 192 floats
      const float* src = (tid < 16) ? (row + h * HEADD + tid * 4)
                                    : (row + DINNER + (tid - 16) * 4);
      const uint64_t gaddr = (uint64_t)(uintptr_t)src;
      const uint32_t laddr = ldsBase + (uint32_t)tid * 16u;
      asm volatile("global_load_async_to_lds_b128 %0, %1, off"
                   :: "v"(laddr), "v"(gaddr) : "memory");
    }
    asm volatile("s_wait_asynccnt 0x0" ::: "memory");
    __syncthreads();
    const float dtl = dt[(size_t)(b * L_SEQ + l) * NH + h];   // uniform broadcast
    const float dA  = __expf(dtl * A);
    const float xp  = stage[p];
    const float xdt = xp * dtl;
    float yp = 0.f;
#pragma unroll
    for (int j = 0; j < 16; ++j) {
      hs[j] = hs[j] * dA + xdt * stage[64 + nb + j];
      yp += hs[j] * stage[128 + nb + j];
    }
    yp += __shfl_xor(yp, 1, 32);
    yp += __shfl_xor(yp, 2, 32);
    if ((tid & 3) == 0)
      y[(size_t)(b * L_SEQ + l) * DINNER + h * HEADD + p] = yp + xp * Dh;
  }
}

// ---------------------------------------------------------------------------
// y = RMSNorm(y * silu(z)) * norm_w   (one block per (b,l) row of 1024)
// ---------------------------------------------------------------------------
__global__ __launch_bounds__(256) void gate_rmsnorm_kernel(const float* __restrict__ zxbcdt,
                                                           const float* __restrict__ nw,
                                                           float* __restrict__ y) {
  const int bl = blockIdx.x;
  const float* z = zxbcdt + (size_t)bl * DPROJ;
  float* yr = y + (size_t)bl * DINNER;
  float g[4];
  float ss = 0.f;
#pragma unroll
  for (int i = 0; i < 4; ++i) {
    const int c = threadIdx.x + i * 256;
    const float zz = z[c];
    const float v = yr[c] * zz / (1.f + __expf(-zz));
    g[i] = v;
    ss += v * v;
  }
  ss = blockReduceSum(ss);
  const float sc = rsqrtf(ss * (1.f / DINNER) + EPSV);
#pragma unroll
  for (int i = 0; i < 4; ++i) {
    const int c = threadIdx.x + i * 256;
    yr[c] = g[i] * sc * nw[c];
  }
}

// ---------------------------------------------------------------------------
// h = LayerNorm(proj + h) * ln_w + ln_b   (one block per (b,l) row of 512)
// ---------------------------------------------------------------------------
__global__ __launch_bounds__(256) void add_layernorm_kernel(const float* __restrict__ proj,
                                                            const float* __restrict__ lw,
                                                            const float* __restrict__ lb,
                                                            float* __restrict__ h) {
  const int bl = blockIdx.x;
  const float* p = proj + (size_t)bl * DM;
  float* hr = h + (size_t)bl * DM;
  float v[2];
  float s = 0.f;
#pragma unroll
  for (int i = 0; i < 2; ++i) {
    const int c = threadIdx.x + i * 256;
    v[i] = p[c] + hr[c];
    s += v[i];
  }
  s = blockReduceSum(s);
  const float mean = s * (1.f / DM);
  float vv = 0.f;
#pragma unroll
  for (int i = 0; i < 2; ++i) {
    const float d = v[i] - mean;
    vv += d * d;
  }
  vv = blockReduceSum(vv);
  const float inv = rsqrtf(vv * (1.f / DM) + EPSV);
#pragma unroll
  for (int i = 0; i < 2; ++i) {
    const int c = threadIdx.x + i * 256;
    hr[c] = (v[i] - mean) * inv * lw[c] + lb[c];
  }
}

// ---------------------------------------------------------------------------
extern "C" void kernel_launch(void* const* d_in, const int* in_sizes, int n_in,
                              void* d_out, int out_size, void* d_ws, size_t ws_size,
                              hipStream_t stream) {
  (void)in_sizes; (void)n_in; (void)out_size; (void)ws_size;
  const float* x          = (const float*)d_in[0];
  const float* in_proj_w  = (const float*)d_in[1];
  const float* conv_w     = (const float*)d_in[2];
  const float* conv_b     = (const float*)d_in[3];
  const float* dt_bias    = (const float*)d_in[4];
  const float* A_log      = (const float*)d_in[5];
  const float* Dp         = (const float*)d_in[6];
  const float* norm_w     = (const float*)d_in[7];
  const float* out_proj_w = (const float*)d_in[8];
  const float* ln_w       = (const float*)d_in[9];
  const float* ln_b       = (const float*)d_in[10];
  const float* final_w    = (const float*)d_in[11];
  const float* final_b    = (const float*)d_in[12];
  float* out = (float*)d_out;

  const size_t BL = (size_t)B_SZ * L_SEQ;            // 16384
  char* ws = (char*)d_ws;
  size_t off = 0;
  auto carve = [&](size_t elems) {
    float* p = (float*)(ws + off);
    off += elems * sizeof(float);
    return p;
  };
  float* zxbcdt = carve(BL * DPROJ);    // 143.7 MB
  float* xbc    = carve(BL * CONVD);    //  75.5 MB
  float* dtbuf  = carve(BL * NH);       //   1.0 MB
  float* ybuf   = carve(BL * DINNER);   //  67.1 MB
  float* hA     = carve(BL * DM);       //  33.6 MB  (running hidden)
  float* hB     = carve(BL * DM);       //  33.6 MB  (out_proj result)

  hipMemcpyAsync(hA, x, BL * DM * sizeof(float), hipMemcpyDeviceToDevice, stream);

  for (int i = 0; i < NL; ++i) {
    // in_proj: zxbcdt = hA @ W_in^T   (blocks: (16384/512) * (2192/16) = 32*137)
    gemm_bf16<<<(16384 / 512) * (DPROJ / 16), 256, 0, stream>>>(
        hA, in_proj_w + (size_t)i * DPROJ * DM, zxbcdt, (int)BL, DPROJ, DM);
    dt_softplus_kernel<<<(B_SZ * L_SEQ * NH) / 256, 256, 0, stream>>>(
        zxbcdt, dt_bias + i * NH, dtbuf);
    conv_silu_kernel<<<(B_SZ * L_SEQ * CONVD) / 256, 256, 0, stream>>>(
        zxbcdt, conv_w + (size_t)i * CONVD * DCONV, conv_b + (size_t)i * CONVD, xbc);
    ssd_scan_kernel<<<B_SZ * NH, 256, 0, stream>>>(
        xbc, dtbuf, A_log + i * NH, Dp + i * NH, ybuf);
    gate_rmsnorm_kernel<<<(int)BL, 256, 0, stream>>>(
        zxbcdt, norm_w + (size_t)i * DINNER, ybuf);
    // out_proj: hB = y @ W_out^T  (blocks: 32*32)
    gemm_bf16<<<(16384 / 512) * (DM / 16), 256, 0, stream>>>(
        ybuf, out_proj_w + (size_t)i * DM * DINNER, hB, (int)BL, DM, DINNER);
    add_layernorm_kernel<<<(int)BL, 256, 0, stream>>>(
        hB, ln_w + (size_t)i * DM, ln_b + (size_t)i * DM, hA);
  }

  // final head on last PRED positions (blocks: (3072/512) * 32 = 6*32)
  gemm_tail_bf16<<<(B_SZ * PRED / 512) * (DM / 16), 256, 0, stream>>>(
      hA, final_w, final_b, out);
}
